// VanillaRNN_13271448945138
// MI455X (gfx1250) — matrix operand order repaired and checked by reading, not compiled, not run on hardware
//
#include <hip/hip_runtime.h>

#define BATCH   512
#define SEQLEN  256
#define HIDDEN  1024
#define CLASSES 128

typedef __attribute__((ext_vector_type(16))) __bf16 bf16x16;
typedef __attribute__((ext_vector_type(8)))  float  f32x8;

// ---------------------------------------------------------------------------
// fp32 -> bf16 (round-to-nearest-even)
// ---------------------------------------------------------------------------
__device__ __forceinline__ unsigned short f32_to_bf16(float f) {
    unsigned int u = __builtin_bit_cast(unsigned int, f);
    u += 0x7FFFu + ((u >> 16) & 1u);
    return (unsigned short)(u >> 16);
}

// ---------------------------------------------------------------------------
// WMMA bf16 fragment loaders (wave32 layouts, 16x16x32)
//
// A (16x32, M x K): lane L<16 holds row M=L, K = [k0..k0+7] in v0..3 and
// [k0+16..k0+23] in v4..7; lanes 16..31 hold the +8 K ranges.
//   -> per-lane base = row_ptr + k0 + 8*hi ; chunks at +0 and +16 elements.
// B (32x16, K x N): lane L<16 holds col N=L, K = [k0..k0+7] and [k0+16..k0+23]; lanes 16..31
// hold the +8 K ranges.  B[k][n] = W[j0+n][k], so a B column is a
// contiguous row of the (row-major) weight matrix.
//   -> per-lane base = row_ptr + k0 + 8*hi ; chunks at +0 and +16 elements.
// ---------------------------------------------------------------------------
__device__ __forceinline__ bf16x16 load_frag_a(const unsigned short* base) {
    union { bf16x16 v; uint4 q[2]; } f;
    f.q[0] = *reinterpret_cast<const uint4*>(base);       // K .. K+7
    f.q[1] = *reinterpret_cast<const uint4*>(base + 16);  // K+16 .. K+23
    return f.v;
}
__device__ __forceinline__ bf16x16 load_frag_b(const unsigned short* base) {
    union { bf16x16 v; uint4 q[2]; } f;
    f.q[0] = *reinterpret_cast<const uint4*>(base);       // K .. K+7
    f.q[1] = *reinterpret_cast<const uint4*>(base + 16);  // K+16 .. K+23
    return f.v;
}

// ---------------------------------------------------------------------------
// Prep kernels
// ---------------------------------------------------------------------------
__global__ void cvt_f32_to_bf16(const float* __restrict__ src,
                                unsigned short* __restrict__ dst, int n) {
    int i = blockIdx.x * blockDim.x + threadIdx.x;
    if (i < n) dst[i] = f32_to_bf16(src[i]);
}

__global__ void zero_u16(unsigned short* __restrict__ dst, int n) {
    int i = blockIdx.x * blockDim.x + threadIdx.x;
    if (i < n) dst[i] = 0;
}

// ---------------------------------------------------------------------------
// One RNN time step:  h_out = tanh( x[:,t] (x) W_hx  +  h_in @ W_hh^T + b_h )
//
// Grid: 64 blocks x 256 threads = 512 waves; each wave owns a 32x32 output
// tile (2x2 WMMA tiles), K-loop over HIDDEN in steps of 32.
//
// Wave->tile mapping: the 8 waves of a workgroup form a 2(M) x 4(N) sub-grid
// of 32x32 tiles, so co-resident waves share A fragments 4-ways and B
// fragments 2-ways through the WGP$ (cuts per-step L2 traffic ~64MB -> ~24MB).
// Block grid: 8 M-groups (of 64 rows) x 8 N-groups (of 128 cols) = 64 blocks.
// ---------------------------------------------------------------------------
__global__ __launch_bounds__(256) void rnn_step_kernel(
    const unsigned short* __restrict__ h_in,   // [B,H] bf16
    const unsigned short* __restrict__ whh,    // [H,H] bf16, row j / col k
    const float*          __restrict__ x,      // [B,SEQLEN] fp32
    const float*          __restrict__ whx,    // [H] fp32
    const float*          __restrict__ bh,     // [H] fp32
    unsigned short*       __restrict__ h_out,  // [B,H] bf16
    int t)
{
    const int wave  = threadIdx.x >> 5;
    const int lane  = threadIdx.x & 31;
    const int mGrp  = blockIdx.x & 7;          // 8 M-groups over BATCH=512
    const int nGrp  = blockIdx.x >> 3;         // 8 N-groups over HIDDEN=1024
    const int tileM = (mGrp * 2 + (wave & 1)) * 32;   // 16 M-blocks total
    const int tileN = (nGrp * 4 + (wave >> 1)) * 32;  // 32 N-blocks total
    const int hi    = lane >> 4;               // half-wave select
    const int lr    = lane & 15;

    f32x8 acc[2][2];
    for (int mi = 0; mi < 2; ++mi)
        for (int ni = 0; ni < 2; ++ni)
            acc[mi][ni] = (f32x8)(0.0f);

    const unsigned short* aRow0 = h_in + (size_t)(tileM + lr)      * HIDDEN + 8  * hi;
    const unsigned short* aRow1 = h_in + (size_t)(tileM + 16 + lr) * HIDDEN + 8  * hi;
    const unsigned short* bRow0 = whh  + (size_t)(tileN + lr)      * HIDDEN + 8  * hi;
    const unsigned short* bRow1 = whh  + (size_t)(tileN + 16 + lr) * HIDDEN + 8  * hi;

    for (int k0 = 0; k0 < HIDDEN; k0 += 32) {
        bf16x16 a0 = load_frag_a(aRow0 + k0);
        bf16x16 a1 = load_frag_a(aRow1 + k0);
        bf16x16 b0 = load_frag_b(bRow0 + k0);
        bf16x16 b1 = load_frag_b(bRow1 + k0);
        acc[0][0] = __builtin_amdgcn_wmma_f32_16x16x32_bf16(false, a0, false, b0,
                        (short)0, acc[0][0], false, false);
        acc[0][1] = __builtin_amdgcn_wmma_f32_16x16x32_bf16(false, a0, false, b1,
                        (short)0, acc[0][1], false, false);
        acc[1][0] = __builtin_amdgcn_wmma_f32_16x16x32_bf16(false, a1, false, b0,
                        (short)0, acc[1][0], false, false);
        acc[1][1] = __builtin_amdgcn_wmma_f32_16x16x32_bf16(false, a1, false, b1,
                        (short)0, acc[1][1], false, false);
    }

    // Epilogue: C/D 16x16 f32 layout -> VGPR r, lanes 0-15: (M=r, N=lane),
    // lanes 16-31: (M=8+r, N=lane-16).
    for (int ni = 0; ni < 2; ++ni) {
        const int j  = tileN + 16 * ni + lr;
        const float wj = whx[j];
        const float bj = bh[j];
        for (int mi = 0; mi < 2; ++mi) {
            const int mBase = tileM + 16 * mi + 8 * hi;
            #pragma unroll
            for (int r = 0; r < 8; ++r) {
                const int m  = mBase + r;
                const float xv = x[(size_t)m * SEQLEN + t];
                const float v  = acc[mi][ni][r] + xv * wj + bj;
                h_out[(size_t)m * HIDDEN + j] = f32_to_bf16(tanhf(v));
            }
        }
    }
}

// ---------------------------------------------------------------------------
// Output head: out = h_last @ W_ph^T + b_p   ([512,1024]x[1024,128] -> fp32)
// Grid: 8 blocks x 256 threads = 64 waves; waves form 2(M)x4(N) sub-grids,
// block grid = 8 M-groups x 1 N-group (CLASSES=128 = 4 tiles of 32).
// ---------------------------------------------------------------------------
__global__ __launch_bounds__(256) void head_kernel(
    const unsigned short* __restrict__ h_last, // [B,H] bf16
    const unsigned short* __restrict__ wph,    // [C,H] bf16
    const float*          __restrict__ bp,     // [C] fp32
    float*                __restrict__ out)    // [B,C] fp32
{
    const int wave  = threadIdx.x >> 5;
    const int lane  = threadIdx.x & 31;
    const int tileM = (blockIdx.x * 2 + (wave & 1)) * 32;  // 16 M-blocks
    const int tileN = (wave >> 1) * 32;                    // 4 N-blocks
    const int hi    = lane >> 4;
    const int lr    = lane & 15;

    f32x8 acc[2][2];
    for (int mi = 0; mi < 2; ++mi)
        for (int ni = 0; ni < 2; ++ni)
            acc[mi][ni] = (f32x8)(0.0f);

    const unsigned short* aRow0 = h_last + (size_t)(tileM + lr)      * HIDDEN + 8  * hi;
    const unsigned short* aRow1 = h_last + (size_t)(tileM + 16 + lr) * HIDDEN + 8  * hi;
    const unsigned short* bRow0 = wph    + (size_t)(tileN + lr)      * HIDDEN + 8  * hi;
    const unsigned short* bRow1 = wph    + (size_t)(tileN + 16 + lr) * HIDDEN + 8  * hi;

    for (int k0 = 0; k0 < HIDDEN; k0 += 32) {
        bf16x16 a0 = load_frag_a(aRow0 + k0);
        bf16x16 a1 = load_frag_a(aRow1 + k0);
        bf16x16 b0 = load_frag_b(bRow0 + k0);
        bf16x16 b1 = load_frag_b(bRow1 + k0);
        acc[0][0] = __builtin_amdgcn_wmma_f32_16x16x32_bf16(false, a0, false, b0,
                        (short)0, acc[0][0], false, false);
        acc[0][1] = __builtin_amdgcn_wmma_f32_16x16x32_bf16(false, a0, false, b1,
                        (short)0, acc[0][1], false, false);
        acc[1][0] = __builtin_amdgcn_wmma_f32_16x16x32_bf16(false, a1, false, b0,
                        (short)0, acc[1][0], false, false);
        acc[1][1] = __builtin_amdgcn_wmma_f32_16x16x32_bf16(false, a1, false, b1,
                        (short)0, acc[1][1], false, false);
    }

    for (int ni = 0; ni < 2; ++ni) {
        const int j  = tileN + 16 * ni + lr;
        const float bj = bp[j];
        for (int mi = 0; mi < 2; ++mi) {
            const int mBase = tileM + 16 * mi + 8 * hi;
            #pragma unroll
            for (int r = 0; r < 8; ++r) {
                const int m = mBase + r;
                out[(size_t)m * CLASSES + j] = acc[mi][ni][r] + bj;
            }
        }
    }
}

// ---------------------------------------------------------------------------
// Host side
// ---------------------------------------------------------------------------
extern "C" void kernel_launch(void* const* d_in, const int* in_sizes, int n_in,
                              void* d_out, int out_size, void* d_ws, size_t ws_size,
                              hipStream_t stream) {
    const float* x   = (const float*)d_in[0];  // [512,256]
    const float* whx = (const float*)d_in[1];  // [1024,1]
    const float* whh = (const float*)d_in[2];  // [1024,1024]
    const float* bh  = (const float*)d_in[3];  // [1024]
    const float* wph = (const float*)d_in[4];  // [128,1024]
    const float* bp  = (const float*)d_in[5];  // [128]
    float* out = (float*)d_out;                // [512,128]

    char* ws = (char*)d_ws;
    unsigned short* whh_bf = (unsigned short*)(ws);                         // 2 MB
    unsigned short* h_a    = (unsigned short*)(ws + (size_t)(2u << 20));    // 1 MB
    unsigned short* h_b    = (unsigned short*)(ws + (size_t)(3u << 20));    // 1 MB
    unsigned short* wph_bf = (unsigned short*)(ws + (size_t)(4u << 20));    // 256 KB

    // Prep: weight conversion (once per call) + zero initial hidden state.
    const int nWhh = HIDDEN * HIDDEN;
    cvt_f32_to_bf16<<<(nWhh + 255) / 256, 256, 0, stream>>>(whh, whh_bf, nWhh);
    const int nWph = CLASSES * HIDDEN;
    cvt_f32_to_bf16<<<(nWph + 255) / 256, 256, 0, stream>>>(wph, wph_bf, nWph);
    const int nH = BATCH * HIDDEN;
    zero_u16<<<(nH + 255) / 256, 256, 0, stream>>>(h_a, nH);

    // Sequential time loop: one WMMA GEMM+tanh kernel per step (graph-captured).
    unsigned short* hin  = h_a;
    unsigned short* hout = h_b;
    for (int t = 0; t < SEQLEN; ++t) {
        rnn_step_kernel<<<64, 256, 0, stream>>>(hin, whh_bf, x, whx, bh, hout, t);
        unsigned short* tmp = hin; hin = hout; hout = tmp;
    }

    // Output head on the final hidden state.
    head_kernel<<<8, 256, 0, stream>>>(hin, wph_bf, bp, out);
}
